// AttentionBlock_79259326481027
// MI455X (gfx1250) — compile-verified, matrix operand order checked by reading, not run on hardware
//
#include <hip/hip_runtime.h>

#define BS   8
#define NSEQ 2048
#define D    128
#define KEXP 2
#define ROWS (BS * NSEQ)

typedef __attribute__((ext_vector_type(16))) __bf16 v16bf;
typedef __attribute__((ext_vector_type(8)))  __bf16 v8bf;
typedef __attribute__((ext_vector_type(8)))  float  v8f;

// 16-bit A/B fragment K index for element e of the v16bf, per CDNA5 ISA
// 16x32 layout: lanes 0-15 hold K {0..7,16..23}, lanes 16-31 hold K {8..15,24..31}.
__device__ __forceinline__ int klane(int e, int half) {
  return ((e & 8) << 1) + half * 8 + (e & 7);
}
__device__ __forceinline__ float lrelu(float x) { return x > 0.f ? x : 0.01f * x; }

// Build a 16-element bf16 fragment from two contiguous 16-byte runs:
// base points at row start + k0; runs at +half*8 and +16+half*8.
__device__ __forceinline__ v16bf load_bfrag(const __bf16* __restrict__ base, int half) {
  const __bf16* p = base + half * 8;
  v8bf lo = *(const v8bf*)(p);
  v8bf hi = *(const v8bf*)(p + 16);
  return __builtin_shufflevector(lo, hi, 0, 1, 2, 3, 4, 5, 6, 7,
                                 8, 9, 10, 11, 12, 13, 14, 15);
}

// ---------------------------------------------------------------- cvt f32->bf16
__global__ void cvt_bf16_kernel(const float* __restrict__ src,
                                __bf16* __restrict__ dst, int n) {
  int i = blockIdx.x * blockDim.x + threadIdx.x;
  if (i < n) dst[i] = (__bf16)src[i];
}

// ------------------------------------------------- generic C = A @ B^T (bf16 WMMA)
// A: [M,K] bf16 row-major, B: [N,K] bf16 row-major, C: [M,N] f32.
// Epilogue options: += addsrc[row,col], += bias[col], bf16 mirror, and a
// per-batch TRANSPOSED bf16 mirror CbfT[(row/NSEQ)*N + col][row%NSEQ]
// (for feeding WMMA B-operands with contiguous loads later).
__global__ void __launch_bounds__(256)
gemm_bf16_xwT(const __bf16* __restrict__ A, const __bf16* __restrict__ B,
              float* __restrict__ C, __bf16* __restrict__ Cbf,
              __bf16* __restrict__ CbfT,
              const float* __restrict__ addsrc, const float* __restrict__ bias,
              int M, int N, int K) {
  int wave = (blockIdx.x * blockDim.x + threadIdx.x) >> 5;
  int lane = threadIdx.x & 31;
  int tilesN = N >> 4;
  int ntiles = (M >> 4) * tilesN;
  if (wave >= ntiles) return;                  // uniform per wave
  int tm = wave / tilesN, tn = wave % tilesN;
  int half = lane >> 4, r = lane & 15;
  int arow = tm * 16 + r;
  int bcol = tn * 16 + r;

  v8f c = {};
  for (int k0 = 0; k0 < K; k0 += 32) {
    v16bf a = load_bfrag(A + (size_t)arow * K + k0, half);
    v16bf b = load_bfrag(B + (size_t)bcol * K + k0, half);
    c = __builtin_amdgcn_wmma_f32_16x16x32_bf16(false, a, false, b,
                                                (short)0, c, false, false);
  }
  int col = tn * 16 + r;
  float vals[8];
#pragma unroll
  for (int v = 0; v < 8; ++v) {
    int row = tm * 16 + v + 8 * half;          // C layout: VGPR v -> M=v / v+8
    float val = c[v];
    if (addsrc) val += addsrc[(size_t)row * N + col];
    if (bias)   val += bias[col];
    C[(size_t)row * N + col] = val;
    vals[v] = val;
    if (Cbf) Cbf[(size_t)row * N + col] = (__bf16)val;
  }
  if (CbfT) {
    v8bf tv;
#pragma unroll
    for (int v = 0; v < 8; ++v) tv[v] = (__bf16)vals[v];
    int row0 = tm * 16 + 8 * half;             // 8 consecutive rows, same batch
    *(v8bf*)(CbfT + ((size_t)((row0 >> 11) * N + col)) * NSEQ + (row0 & (NSEQ - 1))) = tv;
  }
}

// ------------------------------------------------------------- row LayerNorm -> bf16
__global__ void rowln_bf16_kernel(const float* __restrict__ X,
                                  const float* __restrict__ g,
                                  const float* __restrict__ b,
                                  __bf16* __restrict__ out, int rows) {
  int row = blockIdx.x * blockDim.x + threadIdx.x;
  if (row >= rows) return;
  const float* xr = X + (size_t)row * D;
  float m = 0.f;
  for (int c = 0; c < D; ++c) m += xr[c];
  m *= (1.f / D);
  float v = 0.f;
  for (int c = 0; c < D; ++c) { float d = xr[c] - m; v += d * d; }
  v *= (1.f / D);
  float rs = rsqrtf(v + 1e-5f);
  for (int c = 0; c < D; ++c)
    out[(size_t)row * D + c] = (__bf16)((xr[c] - m) * rs * g[c] + b[c]);
}

// ----------------------------------------------------------- s1/s2 row dots
__global__ void s12_kernel(const float* __restrict__ h, const float* __restrict__ aw,
                           float* __restrict__ s1, float* __restrict__ s2, int rows) {
  int row = blockIdx.x * blockDim.x + threadIdx.x;
  if (row >= rows) return;
  const float* hr = h + (size_t)row * D;
  float a = 0.f, b = 0.f;
  for (int c = 0; c < D; ++c) { float v = hr[c]; a += v * aw[c]; b += v * aw[D + c]; }
  s1[row] = a;
  s2[row] = b;
}

// ----------------------------------------------------------- per-batch max of s2
__global__ void s2max_kernel(const float* __restrict__ s2, float* __restrict__ s2m) {
  __shared__ float red[256];
  int b = blockIdx.x;
  float m = -1e30f;
  for (int j = threadIdx.x; j < NSEQ; j += 256) m = fmaxf(m, s2[b * NSEQ + j]);
  red[threadIdx.x] = m;
  __syncthreads();
  for (int s = 128; s > 0; s >>= 1) {
    if (threadIdx.x < s) red[threadIdx.x] = fmaxf(red[threadIdx.x], red[threadIdx.x + s]);
    __syncthreads();
  }
  if (threadIdx.x == 0) s2m[b] = red[0];
}

// ----------------------------------------------------------- softmax denominators
__global__ void denom_kernel(const float* __restrict__ s1, const float* __restrict__ s2,
                             const float* __restrict__ s2m, float* __restrict__ invden) {
  __shared__ float s2t[256];
  int row = blockIdx.x * 256 + threadIdx.x;     // 2048 % 256 == 0 -> batch uniform
  int b = row >> 11;
  float s1v = s1[row];
  float m = lrelu(s1v + s2m[b]);                // exact max (lrelu monotone)
  float den = 0.f;
  for (int j0 = 0; j0 < NSEQ; j0 += 256) {
    __syncthreads();
    s2t[threadIdx.x] = s2[(b << 11) + j0 + threadIdx.x];
    __syncthreads();
    for (int j = 0; j < 256; ++j) den += __expf(lrelu(s1v + s2t[j]) - m);
  }
  invden[row] = 1.f / den;
}

// ---------------------------------------------------------------------------
// Fused attention: per WG = 128 query rows of one batch; barrier-free j-loop.
// B fragments come straight from the transposed hT slab (two 16B loads each,
// 8x reuse inside the WG via WGP$), softmax tile lives in registers, the
// atten_agg row is written once with contiguous 32B runs, P@h accumulates in
// WMMA, then residual + LN2 via LDS and the ff1 GEMM + ELU (WMMA) finish it.
// ---------------------------------------------------------------------------
__global__ void __launch_bounds__(256)
atten_fused(const float* __restrict__ xenc, const __bf16* __restrict__ hbfT,
            const float* __restrict__ s1, const float* __restrict__ s2,
            const float* __restrict__ s2m, const float* __restrict__ invden,
            const float* __restrict__ g2, const float* __restrict__ bb2,
            const __bf16* __restrict__ w1bf,
            float* __restrict__ out_h, float* __restrict__ out_att,
            int accflag, float invk) {
  __shared__ __bf16 attacc[128][D + 8]; // post-residual attentioned (bf16, padded)
  __shared__ float  mrow[128], rsrow[128];

  int b = blockIdx.y;
  int ibase = blockIdx.x * 128;
  int tid = threadIdx.x;
  int wave = tid >> 5, lane = tid & 31, half = lane >> 4, r = lane & 15;
  size_t rowoff = (size_t)b * NSEQ;
  const __bf16* hT = hbfT + (size_t)b * D * NSEQ;   // [d][j] for this batch

  int irow = ibase + wave * 16 + r;     // this lane's query row (A fragment row)
  float s1v = s1[rowoff + irow];
  float mI  = lrelu(s1v + s2m[b]);
  float idn = invden[rowoff + irow];
  const float* s2b = s2 + rowoff;
  float* arow_out = out_att + ((size_t)b * NSEQ + irow) * NSEQ;

  v8f acc[8] = {};                      // 16 rows x 128 d per wave

  for (int j0 = 0; j0 < NSEQ; j0 += 32) {
    // softmax tile (A fragment): two contiguous runs of 8 j's per lane
    float sv[16];
#pragma unroll
    for (int e = 0; e < 8; ++e) sv[e] = s2b[j0 + half * 8 + e];
#pragma unroll
    for (int e = 0; e < 8; ++e) sv[8 + e] = s2b[j0 + 16 + half * 8 + e];

    v16bf a;
    float pv[16];
#pragma unroll
    for (int e = 0; e < 16; ++e) {
      float p = __expf(lrelu(s1v + sv[e]) - mI) * idn;
      pv[e] = p;
      a[e] = (__bf16)p;
    }
    // atten_agg write: contiguous 32B runs (vectorizable), RMW on 2nd expert
    {
      float* ap0 = arow_out + j0 + half * 8;
      float* ap1 = ap0 + 16;
#pragma unroll
      for (int e = 0; e < 8; ++e) {
        float v = pv[e] * invk;
        if (accflag) v += ap0[e];
        ap0[e] = v;
      }
#pragma unroll
      for (int e = 0; e < 8; ++e) {
        float v = pv[8 + e] * invk;
        if (accflag) v += ap1[e];
        ap1[e] = v;
      }
    }
    // acc += P(16x32) @ h(32x128): 8 WMMA tiles over d, B direct from hT
#pragma unroll
    for (int t = 0; t < 8; ++t) {
      v16bf bf = load_bfrag(hT + (size_t)(t * 16 + r) * NSEQ + j0, half);
      acc[t] = __builtin_amdgcn_wmma_f32_16x16x32_bf16(false, a, false, bf,
                                                       (short)0, acc[t], false, false);
    }
  }

  // residual + dump to LDS
#pragma unroll
  for (int t = 0; t < 8; ++t)
#pragma unroll
    for (int v = 0; v < 8; ++v) {
      int rl = wave * 16 + v + 8 * half;
      int col = t * 16 + r;
      attacc[rl][col] = (__bf16)(acc[t][v] + xenc[(rowoff + ibase + rl) * D + col]);
    }
  __syncthreads();

  // LN2 row statistics
  if (tid < 128) {
    float m = 0.f;
    for (int c = 0; c < D; ++c) m += (float)attacc[tid][c];
    m *= (1.f / D);
    float vv = 0.f;
    for (int c = 0; c < D; ++c) { float d = (float)attacc[tid][c] - m; vv += d * d; }
    mrow[tid] = m;
    rsrow[tid] = rsqrtf(vv * (1.f / D) + 1e-5f);
  }
  __syncthreads();

  // h_next tile: elu( LN2(attacc) @ ff1^T ) / k
  int arow = wave * 16 + r;
  float am = mrow[arow], ars = rsrow[arow];
  v8f acc2[8] = {};
  for (int k0 = 0; k0 < D; k0 += 32) {
    v16bf a2;
#pragma unroll
    for (int e = 0; e < 16; ++e) {
      int k = k0 + klane(e, half);
      a2[e] = (__bf16)(((float)attacc[arow][k] - am) * ars * g2[k] + bb2[k]);
    }
#pragma unroll
    for (int t = 0; t < 8; ++t) {
      v16bf b2 = load_bfrag(w1bf + (size_t)(t * 16 + r) * D + k0, half);
      acc2[t] = __builtin_amdgcn_wmma_f32_16x16x32_bf16(false, a2, false, b2,
                                                        (short)0, acc2[t], false, false);
    }
  }
#pragma unroll
  for (int t = 0; t < 8; ++t)
#pragma unroll
    for (int v = 0; v < 8; ++v) {
      int rl = wave * 16 + v + 8 * half;
      int col = t * 16 + r;
      float xv = acc2[t][v];
      float ev = (xv > 0.f ? xv : __expf(xv) - 1.f) * invk;
      size_t idx = (rowoff + ibase + rl) * D + col;
      if (accflag) ev += out_h[idx];
      out_h[idx] = ev;
    }
}

// ---------------------------------------------------------------------------
extern "C" void kernel_launch(void* const* d_in, const int* in_sizes, int n_in,
                              void* d_out, int out_size, void* d_ws, size_t ws_size,
                              hipStream_t stream) {
  (void)in_sizes; (void)n_in; (void)out_size; (void)ws_size;
  const float* x    = (const float*)d_in[0];
  const float* enc  = (const float*)d_in[1];
  const float* wenc = (const float*)d_in[2];
  const float* benc = (const float*)d_in[3];
  const float* ln1g = (const float*)d_in[4];
  const float* ln1b = (const float*)d_in[5];
  const float* ln2g = (const float*)d_in[6];
  const float* ln2b = (const float*)d_in[7];
  const float* ff0w = (const float*)d_in[8];
  const float* ff1w = (const float*)d_in[9];
  const float* attw = (const float*)d_in[10];

  float* out_h   = (float*)d_out;
  float* out_att = out_h + (size_t)ROWS * D;

  char* base = (char*)d_ws;
  size_t off = 0;
  auto alloc = [&](size_t bytes) -> void* {
    void* p = base + off;
    off = (off + bytes + 255) & ~(size_t)255;
    return p;
  };
  __bf16* wenc_bf = (__bf16*)alloc((size_t)D * D * 2);
  __bf16* ff0_bf  = (__bf16*)alloc((size_t)KEXP * D * D * 2);
  __bf16* ff1_bf  = (__bf16*)alloc((size_t)KEXP * D * D * 2);
  __bf16* enc_bf  = (__bf16*)alloc((size_t)ROWS * D * 2);
  float*  xenc    = (float*) alloc((size_t)ROWS * D * 4);
  __bf16* xnorm   = (__bf16*)alloc((size_t)ROWS * D * 2);
  float*  hbuf    = (float*) alloc((size_t)ROWS * D * 4);
  __bf16* hbfT    = (__bf16*)alloc((size_t)ROWS * D * 2);  // per-batch [d][j]
  float*  s1      = (float*) alloc((size_t)ROWS * 4);
  float*  s2      = (float*) alloc((size_t)ROWS * 4);
  float*  invden  = (float*) alloc((size_t)ROWS * 4);
  float*  s2m     = (float*) alloc((size_t)BS * 4);

  cvt_bf16_kernel<<<(D * D + 255) / 256, 256, 0, stream>>>(wenc, wenc_bf, D * D);
  cvt_bf16_kernel<<<(KEXP * D * D + 255) / 256, 256, 0, stream>>>(ff0w, ff0_bf, KEXP * D * D);
  cvt_bf16_kernel<<<(KEXP * D * D + 255) / 256, 256, 0, stream>>>(ff1w, ff1_bf, KEXP * D * D);
  cvt_bf16_kernel<<<(ROWS * D + 255) / 256, 256, 0, stream>>>(enc, enc_bf, ROWS * D);

  int gemm_blocks = (ROWS / 16) * (D / 16) / 8;   // 8 waves / block
  gemm_bf16_xwT<<<gemm_blocks, 256, 0, stream>>>(enc_bf, wenc_bf, xenc, nullptr, nullptr,
                                                 x, benc, ROWS, D, D);
  rowln_bf16_kernel<<<ROWS / 256, 256, 0, stream>>>(xenc, ln1g, ln1b, xnorm, ROWS);

  for (int i = 0; i < KEXP; ++i) {
    gemm_bf16_xwT<<<gemm_blocks, 256, 0, stream>>>(xnorm, ff0_bf + (size_t)i * D * D,
                                                   hbuf, nullptr, hbfT,
                                                   nullptr, nullptr, ROWS, D, D);
    s12_kernel<<<ROWS / 256, 256, 0, stream>>>(hbuf, attw + (size_t)i * 2 * D, s1, s2, ROWS);
    s2max_kernel<<<BS, 256, 0, stream>>>(s2, s2m);
    denom_kernel<<<ROWS / 256, 256, 0, stream>>>(s1, s2, s2m, invden);
    atten_fused<<<dim3(NSEQ / 128, BS), 256, 0, stream>>>(
        xenc, hbfT, s1, s2, s2m, invden, ln2g, ln2b,
        ff1_bf + (size_t)i * D * D, out_h, out_att, i, 1.0f / KEXP);
  }
}
